// PathAndCycleNet_61838939128054
// MI455X (gfx1250) — matrix operand order repaired and checked by reading, not compiled, not run on hardware
//
#include <hip/hip_runtime.h>

typedef __bf16 bf16;
typedef __attribute__((ext_vector_type(16))) __bf16 v16bf;
typedef __attribute__((ext_vector_type(8)))  __bf16 v8bf;
typedef __attribute__((ext_vector_type(8)))  float  v8f;

constexpr int HD      = 64;
constexpr int NATOMS  = 100000;
constexpr int NPATH   = 100000;
constexpr int NCYC    = 40000;
constexpr int BSZ     = 4096;

__device__ __forceinline__ bf16 f2bf(float f) { return (bf16)f; }

union V16U { v16bf v; v8bf h[2]; };

// load a 16-element bf16 A-fragment from two contiguous 16B runs: p[0..7], p[16..23]
__device__ __forceinline__ v16bf load_frag(const bf16* p) {
    V16U u;
    u.h[0] = *(const v8bf*)p;
    u.h[1] = *(const v8bf*)(p + 16);
    return u.v;
}

// ---- swizzled weight fill: B-fragment layout for v_wmma_f32_16x16x32_bf16 ----
// For chunk (kc,nb), lane L (half=L>>4, n=L&15), element j:  B[K= kc*32+(j>>3)*16+half*8+(j&7)][N= nb*16+n]
__device__ void fill_w64(bf16* lds, const float* __restrict__ W, int tid, int nthr) {
    for (int idx = tid; idx < 4096; idx += nthr) {          // 2 kc * 4 nb * 32 lanes * 16
        int j = idx & 15, lane = (idx >> 4) & 31, nb = (idx >> 9) & 3, kc = idx >> 11;
        int hf = lane >> 4, n = lane & 15;
        int K = kc * 32 + ((j >> 3) << 4) + hf * 8 + (j & 7);
        lds[idx] = f2bf(W[K * HD + nb * 16 + n]);
    }
}

// conv kernel [3][64][64] flattened to logical 192-K GEMM; cyclic => tap-symmetrized
__device__ void fill_conv(bf16* lds, const float* __restrict__ K3, bool cyclic, int tid, int nthr) {
    for (int idx = tid; idx < 12288; idx += nthr) {         // 6 kc * 4 nb * 32 * 16
        int j = idx & 15, lane = (idx >> 4) & 31, nb = (idx >> 9) & 3, kc = idx >> 11;
        int hf = lane >> 4, n = lane & 15;
        int Kg = kc * 32 + ((j >> 3) << 4) + hf * 8 + (j & 7); // 0..191
        int t = Kg >> 6, kr = Kg & 63, col = nb * 16 + n;
        float v = K3[(t * HD + kr) * HD + col];
        if (cyclic) v = 0.5f * (v + K3[((2 - t) * HD + kr) * HD + col]);
        lds[idx] = f2bf(v);
    }
}

// ---------------- init kernels ----------------
__global__ void atom_init_k(const int* __restrict__ xa,
                            const float* __restrict__ e0, const float* __restrict__ e1,
                            const float* __restrict__ e2,
                            float* __restrict__ x, bf16* __restrict__ xb) {
    long i = (long)blockIdx.x * blockDim.x + threadIdx.x;
    if (i >= (long)NATOMS * HD) return;
    int a = (int)(i >> 6), h = (int)(i & 63);
    float v = e0[xa[a * 3 + 0] * HD + h] + e1[xa[a * 3 + 1] * HD + h] + e2[xa[a * 3 + 2] * HD + h];
    x[i] = v;
    xb[i] = f2bf(v);
}

__global__ void path_init_k(const int* __restrict__ v, const float* __restrict__ emb,
                            bf16* __restrict__ xp, long n64) {
    long i = (long)blockIdx.x * blockDim.x + threadIdx.x;
    if (i >= n64) return;
    int e = (int)(i >> 6), h = (int)(i & 63);
    xp[i] = f2bf(emb[v[e] * HD + h]);
}

__global__ void zero_k(float* __restrict__ p, long n) {
    long i = (long)blockIdx.x * blockDim.x + threadIdx.x;
    if (i < n) p[i] = 0.0f;
}

// ---------------- fused a2p GEMM + residual conv block (one wave = 16 paths) ----------------
template <int KLEN, bool CYCLIC>
__global__ __launch_bounds__(64) void pc_block_k(
    const bf16* __restrict__ xb,     // [NATOMS*64] atom states (bf16 mirror)
    bf16*       __restrict__ xp,     // [E*64] path state (bf16), updated in place
    const int*  __restrict__ row,    // [E]
    const float* __restrict__ Wa, const float* __restrict__ ba,
    const float* __restrict__ Kc, const float* __restrict__ Kb) {
    __shared__ __attribute__((aligned(32))) bf16 sWa[4096];
    __shared__ __attribute__((aligned(32))) bf16 sK[12288];
    __shared__ float stage[2][16 * KLEN * HD];
    __shared__ __attribute__((aligned(32))) bf16 stageb[2][16 * KLEN * HD];

    int tid = threadIdx.x;
    fill_w64(sWa, Wa, tid, 64);
    fill_conv(sK, Kc, CYCLIC, tid, 64);
    __syncthreads();

    int wave = tid >> 5, lane = tid & 31;
    int hf = lane >> 4, lm = lane & 15;
    long pbase = (long)blockIdx.x * 32 + wave * 16;
    long ebase = pbase * KLEN;
    float* st = stage[wave];
    bf16*  sb = stageb[wave];

    // ---- stage 1: h = xp_old + relu(xb[row] @ Wa + ba) ----
    for (int mt = 0; mt < KLEN; ++mt) {
        int e = (int)ebase + mt * 16 + lm;                  // this lane's A row
        const bf16* xr = xb + (long)row[e] * HD;
        // chunk0: K in {hf*8..+7, 16+hf*8..+7}; chunk1: +32
        v16bf a0 = load_frag(xr + hf * 8);
        v16bf a1 = load_frag(xr + 32 + hf * 8);
        v8f acc[4];
#pragma unroll
        for (int nb = 0; nb < 4; ++nb) {
            v8f c = {};
            v16bf b0 = *(const v16bf*)&sWa[(0 * 4 + nb) * 512 + lane * 16];
            c = __builtin_amdgcn_wmma_f32_16x16x32_bf16(false, a0, false, b0, (short)0, c, false, false);
            v16bf b1 = *(const v16bf*)&sWa[(1 * 4 + nb) * 512 + lane * 16];
            c = __builtin_amdgcn_wmma_f32_16x16x32_bf16(false, a1, false, b1, (short)0, c, false, false);
            acc[nb] = c;
        }
#pragma unroll
        for (int nb = 0; nb < 4; ++nb) {
            int N = nb * 16 + lm;
            float bias = ba[N];
#pragma unroll
            for (int r = 0; r < 8; ++r) {
                int M = r + 8 * hf;
                int me = mt * 16 + M;
                float v = acc[nb][r] + bias;
                v = v > 0.f ? v : 0.f;
                float hv = (float)xp[(ebase + me) * HD + N] + v;
                st[me * HD + N] = hv;
                sb[me * HD + N] = f2bf(hv);
            }
        }
    }

    // ---- stage 2: out = h + relu(conv3(h) + Kb), width-3, SAME / circular ----
    for (int mt = 0; mt < KLEN; ++mt) {
        int me = mt * 16 + lm;                              // output row for this lane's A
        int p = me / KLEN, js = me % KLEN;
        v8f acc[4] = {{}, {}, {}, {}};
#pragma unroll
        for (int kc = 0; kc < 6; ++kc) {
            int t = kc >> 1, kh = kc & 1;
            int s = js + t - 1;
            bool valid = true;
            if (CYCLIC) s = (s + KLEN) % KLEN;
            else        valid = (s >= 0 && s < KLEN);
            v16bf a;
            if (valid) {
                a = load_frag(&sb[(p * KLEN + s) * HD + kh * 32 + hf * 8]);
            } else {
                a = (v16bf){};
            }
#pragma unroll
            for (int nb = 0; nb < 4; ++nb) {
                v16bf b = *(const v16bf*)&sK[(kc * 4 + nb) * 512 + lane * 16];
                acc[nb] = __builtin_amdgcn_wmma_f32_16x16x32_bf16(false, a, false, b, (short)0, acc[nb], false, false);
            }
        }
#pragma unroll
        for (int nb = 0; nb < 4; ++nb) {
            int N = nb * 16 + lm;
            float bias = Kb[N];
#pragma unroll
            for (int r = 0; r < 8; ++r) {
                int M = r + 8 * hf;
                int me2 = mt * 16 + M;
                float c = acc[nb][r] + bias;
                c = c > 0.f ? c : 0.f;
                xp[(ebase + me2) * HD + N] = f2bf(st[me2 * HD + N] + c);
            }
        }
    }
}

// ---------------- scatter (segment sum + count), fp32 source ----------------
__global__ void scatter_k(const float* __restrict__ src, const int* __restrict__ row,
                          float* __restrict__ sum, float* __restrict__ cnt, long E) {
    long i = (long)blockIdx.x * blockDim.x + threadIdx.x;
    if (i >= E * HD) return;
    int e = (int)(i >> 6), h = (int)(i & 63);
    int a = row[e];
    atomicAdd(&sum[(long)a * HD + h], src[i]);
    if (h == 0) atomicAdd(&cnt[a], 1.0f);
}

// ---------------- scatter (segment sum + count), bf16 source, 8 features/thread ----------------
__global__ void scatter_bf_k(const bf16* __restrict__ src, const int* __restrict__ row,
                             float* __restrict__ sum, float* __restrict__ cnt, long E) {
    long i = (long)blockIdx.x * blockDim.x + threadIdx.x;   // over E*8
    if (i >= E * 8) return;
    int e = (int)(i >> 3), h0 = (int)(i & 7) * 8;
    int a = row[e];
    const unsigned* s = (const unsigned*)(src + (long)e * HD + h0);
    float* dp = sum + (long)a * HD + h0;
#pragma unroll
    for (int q = 0; q < 4; ++q) {
        unsigned u = s[q];
        atomicAdd(dp + 2 * q,     __builtin_bit_cast(float, u << 16));        // low bf16
        atomicAdd(dp + 2 * q + 1, __builtin_bit_cast(float, u & 0xFFFF0000u)); // high bf16
    }
    if (h0 == 0) atomicAdd(&cnt[a], 1.0f);
}

// ---------------- p2a: x += relu((sum/max(cnt,1)) @ Wp + bp), refresh bf16 mirror ----------------
__global__ __launch_bounds__(64) void p2a_gemm_k(
    float* __restrict__ x, bf16* __restrict__ xb,
    const float* __restrict__ sum, const float* __restrict__ cnt,
    const float* __restrict__ Wp, const float* __restrict__ bp) {
    __shared__ __attribute__((aligned(32))) bf16 sW[4096];
    int tid = threadIdx.x;
    fill_w64(sW, Wp, tid, 64);
    __syncthreads();
    int wave = tid >> 5, lane = tid & 31;
    int hf = lane >> 4, lm = lane & 15;
    long abase = (long)blockIdx.x * 32 + wave * 16;
    int am = (int)abase + lm;
    float rc = 1.0f / fmaxf(cnt[am], 1.0f);
    const float* sr = sum + (long)am * HD;
    v16bf a0, a1;
#pragma unroll
    for (int j = 0; j < 16; ++j) {
        int K32 = ((j >= 8) ? 16 : 0) + hf * 8 + (j & 7);
        a0[j] = f2bf(sr[K32] * rc);
        a1[j] = f2bf(sr[32 + K32] * rc);
    }
    v8f acc[4];
#pragma unroll
    for (int nb = 0; nb < 4; ++nb) {
        v8f c = {};
        v16bf b0 = *(const v16bf*)&sW[(0 * 4 + nb) * 512 + lane * 16];
        c = __builtin_amdgcn_wmma_f32_16x16x32_bf16(false, a0, false, b0, (short)0, c, false, false);
        v16bf b1 = *(const v16bf*)&sW[(1 * 4 + nb) * 512 + lane * 16];
        c = __builtin_amdgcn_wmma_f32_16x16x32_bf16(false, a1, false, b1, (short)0, c, false, false);
        acc[nb] = c;
    }
#pragma unroll
    for (int nb = 0; nb < 4; ++nb) {
        int N = nb * 16 + lm;
        float bias = bp[N];
#pragma unroll
        for (int r = 0; r < 8; ++r) {
            int M = r + 8 * hf;
            float v = acc[nb][r] + bias;
            v = v > 0.f ? v : 0.f;
            long xi = (abase + M) * HD + N;
            float nv = x[xi] + v;
            x[xi] = nv;
            xb[xi] = f2bf(nv);
        }
    }
}

// ---------------- final: g = relu(mean @ W + b) ----------------
__global__ __launch_bounds__(64) void final_gemm_k(
    float* __restrict__ g, const float* __restrict__ gsum, const float* __restrict__ gcnt,
    const float* __restrict__ W, const float* __restrict__ b) {
    __shared__ __attribute__((aligned(32))) bf16 sW[4096];
    int tid = threadIdx.x;
    fill_w64(sW, W, tid, 64);
    __syncthreads();
    int wave = tid >> 5, lane = tid & 31;
    int hf = lane >> 4, lm = lane & 15;
    long abase = (long)blockIdx.x * 32 + wave * 16;
    int am = (int)abase + lm;
    float rc = 1.0f / fmaxf(gcnt[am], 1.0f);
    const float* sr = gsum + (long)am * HD;
    v16bf a0, a1;
#pragma unroll
    for (int j = 0; j < 16; ++j) {
        int K32 = ((j >= 8) ? 16 : 0) + hf * 8 + (j & 7);
        a0[j] = f2bf(sr[K32] * rc);
        a1[j] = f2bf(sr[32 + K32] * rc);
    }
    v8f acc[4];
#pragma unroll
    for (int nb = 0; nb < 4; ++nb) {
        v8f c = {};
        v16bf b0 = *(const v16bf*)&sW[(0 * 4 + nb) * 512 + lane * 16];
        c = __builtin_amdgcn_wmma_f32_16x16x32_bf16(false, a0, false, b0, (short)0, c, false, false);
        v16bf b1 = *(const v16bf*)&sW[(1 * 4 + nb) * 512 + lane * 16];
        c = __builtin_amdgcn_wmma_f32_16x16x32_bf16(false, a1, false, b1, (short)0, c, false, false);
        acc[nb] = c;
    }
#pragma unroll
    for (int nb = 0; nb < 4; ++nb) {
        int N = nb * 16 + lm;
        float bias = b[N];
#pragma unroll
        for (int r = 0; r < 8; ++r) {
            int M = r + 8 * hf;
            float v = acc[nb][r] + bias;
            g[(abase + M) * HD + N] = v > 0.f ? v : 0.f;
        }
    }
}

__global__ void final_out_k(const float* __restrict__ g, const float* __restrict__ linW,
                            const float* __restrict__ linb, float* __restrict__ out) {
    int i = blockIdx.x * blockDim.x + threadIdx.x;
    if (i >= BSZ) return;
    float s = linb[0];
#pragma unroll
    for (int h = 0; h < HD; ++h) s += g[(long)i * HD + h] * linW[h];
    out[i] = s;
}

// =====================================================================
extern "C" void kernel_launch(void* const* d_in, const int* in_sizes, int n_in,
                              void* d_out, int out_size, void* d_ws, size_t ws_size,
                              hipStream_t stream) {
    (void)in_sizes; (void)n_in; (void)out_size; (void)ws_size;
    // ---- inputs (setup_inputs dict order) ----
    const int* x_atom = (const int*)d_in[0];
    const int* xpv[3]  = {(const int*)d_in[1], (const int*)d_in[4], (const int*)d_in[7]};
    const int* prow[3] = {(const int*)d_in[2], (const int*)d_in[5], (const int*)d_in[8]};
    const int* xcv[2]  = {(const int*)d_in[10], (const int*)d_in[13]};
    const int* crow[2] = {(const int*)d_in[11], (const int*)d_in[14]};
    const int* batch   = (const int*)d_in[16];
    const float* aemb0 = (const float*)d_in[17];
    const float* aemb1 = (const float*)d_in[18];
    const float* aemb2 = (const float*)d_in[19];
    const float* pemb  = (const float*)d_in[20];   // [3][6][64]
    const float* cemb  = (const float*)d_in[21];   // [2][4][64]
    const float* pWa2p = (const float*)d_in[22];   // [2][3][64][64]
    const float* pba2p = (const float*)d_in[23];
    const float* pWp2a = (const float*)d_in[24];
    const float* pbp2a = (const float*)d_in[25];
    const float* pK    = (const float*)d_in[26];   // [2][3][3][64][64]
    const float* pKb   = (const float*)d_in[27];
    const float* cWa2p = (const float*)d_in[28];   // [2][2][64][64]
    const float* cba2p = (const float*)d_in[29];
    const float* cWp2a = (const float*)d_in[30];
    const float* cbp2a = (const float*)d_in[31];
    const float* cK    = (const float*)d_in[32];   // [2][2][3][64][64]
    const float* cKb   = (const float*)d_in[33];
    const float* alW   = (const float*)d_in[34];
    const float* alb   = (const float*)d_in[35];
    const float* linW  = (const float*)d_in[36];
    const float* linb  = (const float*)d_in[37];
    float* out = (float*)d_out;

    const long EP[3] = {(long)NPATH * 3, (long)NPATH * 4, (long)NPATH * 5};
    const long EC[2] = {(long)NCYC * 5, (long)NCYC * 6};

    // ---- workspace layout ----
    char* wsp = (char*)d_ws;
    size_t off = 0;
    auto take = [&](size_t bytes) { void* p = wsp + off; off = (off + bytes + 255) & ~(size_t)255; return p; };
    float* x  = (float*)take((size_t)NATOMS * HD * 4);
    bf16*  xb = (bf16*)take((size_t)NATOMS * HD * 2);
    bf16* pst[3]; for (int i = 0; i < 3; ++i) pst[i] = (bf16*)take((size_t)EP[i] * HD * 2);
    bf16* cst[2]; for (int i = 0; i < 2; ++i) cst[i] = (bf16*)take((size_t)EC[i] * HD * 2);
    float* sumcnt = (float*)take(((size_t)NATOMS * HD + NATOMS) * 4);
    float* sum = sumcnt; float* cnt = sumcnt + (size_t)NATOMS * HD;
    float* gbuf = (float*)take(((size_t)BSZ * HD + BSZ) * 4);
    float* gsum = gbuf; float* gcnt = gbuf + (size_t)BSZ * HD;
    float* g = (float*)take((size_t)BSZ * HD * 4);

    const long sumcnt_n = (long)NATOMS * HD + NATOMS;
    const long gbuf_n   = (long)BSZ * HD + BSZ;

    // ---- init embeddings ----
    {
        long n = (long)NATOMS * HD;
        atom_init_k<<<(int)((n + 255) / 256), 256, 0, stream>>>(x_atom, aemb0, aemb1, aemb2, x, xb);
        for (int i = 0; i < 3; ++i) {
            long n64 = EP[i] * HD;
            path_init_k<<<(int)((n64 + 255) / 256), 256, 0, stream>>>(xpv[i], pemb + i * 6 * HD, pst[i], n64);
        }
        for (int i = 0; i < 2; ++i) {
            long n64 = EC[i] * HD;
            path_init_k<<<(int)((n64 + 255) / 256), 256, 0, stream>>>(xcv[i], cemb + i * 4 * HD, cst[i], n64);
        }
    }

    for (int l = 0; l < 2; ++l) {
        // ===== path conv layer (non-cyclic) =====
        const int gp = NPATH / 32;
        pc_block_k<3, false><<<gp, 64, 0, stream>>>(xb, pst[0], prow[0],
            pWa2p + (l * 3 + 0) * 4096, pba2p + (l * 3 + 0) * 64,
            pK + (size_t)(l * 3 + 0) * 3 * 4096, pKb + (l * 3 + 0) * 64);
        pc_block_k<4, false><<<gp, 64, 0, stream>>>(xb, pst[1], prow[1],
            pWa2p + (l * 3 + 1) * 4096, pba2p + (l * 3 + 1) * 64,
            pK + (size_t)(l * 3 + 1) * 3 * 4096, pKb + (l * 3 + 1) * 64);
        pc_block_k<5, false><<<gp, 64, 0, stream>>>(xb, pst[2], prow[2],
            pWa2p + (l * 3 + 2) * 4096, pba2p + (l * 3 + 2) * 64,
            pK + (size_t)(l * 3 + 2) * 3 * 4096, pKb + (l * 3 + 2) * 64);
        for (int i = 0; i < 3; ++i) {
            zero_k<<<(int)((sumcnt_n + 255) / 256), 256, 0, stream>>>(sum, sumcnt_n);
            long n8 = EP[i] * 8;
            scatter_bf_k<<<(int)((n8 + 255) / 256), 256, 0, stream>>>(pst[i], prow[i], sum, cnt, EP[i]);
            p2a_gemm_k<<<NATOMS / 32, 64, 0, stream>>>(x, xb, sum, cnt,
                pWp2a + (l * 3 + i) * 4096, pbp2a + (l * 3 + i) * 64);
        }
        // ===== cycle conv layer (cyclic, symmetric kernel) =====
        const int gc = NCYC / 32;
        pc_block_k<5, true><<<gc, 64, 0, stream>>>(xb, cst[0], crow[0],
            cWa2p + (l * 2 + 0) * 4096, cba2p + (l * 2 + 0) * 64,
            cK + (size_t)(l * 2 + 0) * 3 * 4096, cKb + (l * 2 + 0) * 64);
        pc_block_k<6, true><<<gc, 64, 0, stream>>>(xb, cst[1], crow[1],
            cWa2p + (l * 2 + 1) * 4096, cba2p + (l * 2 + 1) * 64,
            cK + (size_t)(l * 2 + 1) * 3 * 4096, cKb + (l * 2 + 1) * 64);
        for (int i = 0; i < 2; ++i) {
            zero_k<<<(int)((sumcnt_n + 255) / 256), 256, 0, stream>>>(sum, sumcnt_n);
            long n8 = EC[i] * 8;
            scatter_bf_k<<<(int)((n8 + 255) / 256), 256, 0, stream>>>(cst[i], crow[i], sum, cnt, EC[i]);
            p2a_gemm_k<<<NATOMS / 32, 64, 0, stream>>>(x, xb, sum, cnt,
                cWp2a + (l * 2 + i) * 4096, cbp2a + (l * 2 + i) * 64);
        }
    }

    // ===== readout =====
    zero_k<<<(int)((gbuf_n + 255) / 256), 256, 0, stream>>>(gbuf, gbuf_n);
    {
        long n64 = (long)NATOMS * HD;
        scatter_k<<<(int)((n64 + 255) / 256), 256, 0, stream>>>(x, batch, gsum, gcnt, NATOMS);
    }
    final_gemm_k<<<BSZ / 32, 64, 0, stream>>>(g, gsum, gcnt, alW, alb);
    final_out_k<<<(BSZ + 255) / 256, 256, 0, stream>>>(g, linW, linb, out);
}
